// Decoder_56349970923725
// MI455X (gfx1250) — matrix operand order repaired and checked by reading, not compiled, not run on hardware
//
#include <hip/hip_runtime.h>
#include <hip/hip_bf16.h>
#include <math.h>

typedef __attribute__((ext_vector_type(16))) __bf16 v16bf;
typedef __attribute__((ext_vector_type(8)))  __bf16 v8bf;
typedef __attribute__((ext_vector_type(8)))  float  v8f;

// ------------------------- elementwise / graph kernels -------------------------

__global__ void k_fill(float* __restrict__ p, long long n, float v) {
  long long i = (long long)blockIdx.x * blockDim.x + threadIdx.x;
  if (i < n) p[i] = v;
}

__global__ void k_deg_accum(const int* __restrict__ dst, float* __restrict__ deg, int E) {
  int e = blockIdx.x * blockDim.x + threadIdx.x;
  if (e < E) atomicAdd(&deg[dst[e]], 1.0f);
}

__global__ void k_rsqrt(float* __restrict__ p, long long n) {
  long long i = (long long)blockIdx.x * blockDim.x + threadIdx.x;
  if (i < n) p[i] = rsqrtf(p[i]);
}

__global__ void k_cvt_bf16(const float* __restrict__ in, __bf16* __restrict__ out, long long n) {
  long long i = (long long)blockIdx.x * blockDim.x + threadIdx.x;
  if (i < n) out[i] = (__bf16)in[i];
}

// Convert nmats row-major [K,Nc] f32 matrices into bf16 transposed [Nc,K] matrices.
__global__ void k_cvt_bf16_T(const float* __restrict__ W, __bf16* __restrict__ WT,
                             int K, int Nc, long long total) {
  long long i = (long long)blockIdx.x * blockDim.x + threadIdx.x;
  if (i >= total) return;
  long long mat = i / ((long long)K * Nc);
  long long r   = i - mat * (long long)K * Nc;
  int k = (int)(r / Nc);
  int n = (int)(r - (long long)k * Nc);
  WT[mat * (long long)K * Nc + (size_t)n * K + k] = (__bf16)W[i];
}

// pack a [N,128] f32 matrix into columns [coloff, coloff+128) of a bf16 [N,256] matrix
__global__ void k_pack(const float* __restrict__ in, __bf16* __restrict__ comb,
                       long long n, int coloff) {
  long long i = (long long)blockIdx.x * blockDim.x + threadIdx.x;
  if (i >= n) return;
  long long r = i >> 7;
  int c = (int)(i & 127);
  comb[r * 256 + coloff + c] = (__bf16)in[i];
}

__global__ void k_edge_w(const int* __restrict__ src, const int* __restrict__ dst,
                         const float* __restrict__ dinv, float* __restrict__ w, int E) {
  int e = blockIdx.x * blockDim.x + threadIdx.x;
  if (e < E) w[e] = dinv[src[e]] * dinv[dst[e]];
}

// out[i,c] = dinv[i]^2 * xw[i,c] + bias[c]
__global__ void k_self_bias(const float* __restrict__ xw, const float* __restrict__ dinv,
                            const float* __restrict__ bias, float* __restrict__ out,
                            long long n, int C) {
  long long i = (long long)blockIdx.x * blockDim.x + threadIdx.x;
  if (i >= n) return;
  long long r = i / C;
  int c = (int)(i - r * C);
  float di = dinv[r];
  out[i] = di * di * xw[i] + bias[c];
}

// agg[dst[e], c] += xw[src[e], c] * ew[e]   (channels fastest across lanes, float4)
__global__ void k_edge_scatter(const int* __restrict__ src, const int* __restrict__ dst,
                               const float* __restrict__ ew, const float* __restrict__ xw,
                               float* __restrict__ agg, long long total, int C) {
  long long t = (long long)blockIdx.x * blockDim.x + threadIdx.x;
  if (t >= total) return;
  int cper = C >> 2;
  int e  = (int)(t / cper);
  int c4 = (int)(t - (long long)e * cper) << 2;
  float w = ew[e];
  const float4 v = *(const float4*)(xw + (size_t)src[e] * C + c4);
  float* o = agg + (size_t)dst[e] * C + c4;
  atomicAdd(o + 0, v.x * w);
  atomicAdd(o + 1, v.y * w);
  atomicAdd(o + 2, v.z * w);
  atomicAdd(o + 3, v.w * w);
}

__device__ __forceinline__ float sigmoidf_(float x) { return 1.0f / (1.0f + expf(-x)); }

// LSTM gates: agg is [N,512] (f,i,o,g blocks of 128); updates cbuf/hbuf [N,128], writes hs slice
__global__ void k_lstm(const float* __restrict__ agg, float* __restrict__ cbuf,
                       float* __restrict__ hbuf, float* __restrict__ hs, long long n) {
  long long i = (long long)blockIdx.x * blockDim.x + threadIdx.x;
  if (i >= n) return;
  long long r = i >> 7;
  int c = (int)(i & 127);
  const float* row = agg + (size_t)r * 512;
  float f  = sigmoidf_(row[c]);
  float ig = sigmoidf_(row[128 + c]);
  float o  = sigmoidf_(row[256 + c]);
  float g  = tanhf(row[384 + c]);
  float cn = f * cbuf[i] + ig * g;
  float hn = o * tanhf(cn);
  cbuf[i] = cn;
  hbuf[i] = hn;
  hs[i]   = hn;
}

__global__ void k_copy(const float* __restrict__ in, float* __restrict__ out, long long n) {
  long long i = (long long)blockIdx.x * blockDim.x + threadIdx.x;
  if (i < n) out[i] = in[i];
}

// ------------------------- WMMA bf16 GEMM -------------------------
// C[M,Nc] = A[M,K] @ B[K,Nc] with B provided TRANSPOSED (BT is [Nc,K] row-major),
// f32 accumulate. One wave owns a 16x16 C tile; each wave caches all KT
// B-fragments in registers (now contiguous 32B loads) and streams MT m-tiles.
// A/B fragments use contiguous k = 16*(lane/16) + elem, same for both (wave32).
template <int KT, int MT>
__global__ __launch_bounds__(256) void wmma_gemm_bf16(
    const __bf16* __restrict__ A, int lda,
    const __bf16* __restrict__ BT, int ldbt /* == K */,
    float* __restrict__ C, int ldc, int Mtiles) {
  const int lane  = threadIdx.x & 31;
  const int wave  = threadIdx.x >> 5;
  const int nt    = blockIdx.y;
  const int lhalf = lane & 15;

  // B fragments: lane holds column nb; lanes 0-15 -> K rows kr0..kr0+15 (contiguous in BT)
  v16bf bf[KT];
  {
    const int nb  = nt * 16 + lhalf;
    const int kr0 = (lane < 16) ? 0 : 16;
    const __bf16* bp = BT + (size_t)nb * ldbt + kr0;
    #pragma unroll
    for (int kt = 0; kt < KT; ++kt)
      bf[kt] = *(const v16bf*)(bp + kt * 32);
  }

  union AV { v16bf v; v8bf h[2]; };

  const int mt0 = (blockIdx.x * 8 + wave) * MT;
  const int kc0 = (lane < 16) ? 0 : 16;
  #pragma unroll
  for (int jm = 0; jm < MT; ++jm) {
    const int mt = mt0 + jm;
    if (mt >= Mtiles) break;  // wave-uniform: EXEC stays all-ones for WMMA
    const int ma = mt * 16 + lhalf;
    const __bf16* arow = A + (size_t)ma * lda + kc0;
    v8f acc = {};
    #pragma unroll
    for (int kt = 0; kt < KT; ++kt) {
      // A frag: elems 0..15 = K {kc0..kc0+15}
      AV a;
      a.h[0] = *(const v8bf*)(arow + kt * 32);
      a.h[1] = *(const v8bf*)(arow + kt * 32 + 8);
      acc = __builtin_amdgcn_wmma_f32_16x16x32_bf16(
          /*neg_a=*/false, a.v, /*neg_b=*/false, bf[kt],
          /*c_mod=*/(short)0, acc, /*reuse_a=*/false, /*reuse_b=*/false);
    }
    // C frag: VGPR v -> row (mt*16 + v) lanes 0-15, (mt*16 + 8 + v) lanes 16-31
    const int cn = nt * 16 + lhalf;
    const int rb = mt * 16 + ((lane < 16) ? 0 : 8);
    float* crow = C + (size_t)rb * ldc + cn;
    #pragma unroll
    for (int v = 0; v < 8; ++v) crow[(size_t)v * ldc] = acc[v];
  }
}

// ------------------------- driver -------------------------

static inline unsigned g1(long long n) { return (unsigned)((n + 255) / 256); }

extern "C" void kernel_launch(void* const* d_in, const int* in_sizes, int n_in,
                              void* d_out, int out_size, void* d_ws, size_t ws_size,
                              hipStream_t stream) {
  const float* x     = (const float*)d_in[0];
  const float* c_in  = (const float*)d_in[1];
  const int*   ei    = (const int*)d_in[2];
  const float* Wh    = (const float*)d_in[3];
  const float* bh    = (const float*)d_in[4];
  const float* Wc    = (const float*)d_in[5];
  const float* bc    = (const float*)d_in[6];
  const float* Wcell = (const float*)d_in[7];
  const float* bcell = (const float*)d_in[8];
  float* out = (float*)d_out;

  const int H = 128, O = 128;
  const int N  = in_sizes[0] / H;
  const int E  = in_sizes[2] / 2;
  const int K2 = H + O;                 // 256
  const int C4 = 4 * O;                 // 512
  const int S  = in_sizes[7] / (K2 * C4);
  const long long NH = (long long)N * H;

  const int* srcIdx = ei;
  const int* dstIdx = ei + E;

  // ---- workspace carve-out ----
  char* ws = (char*)d_ws;
  size_t off = 0;
  auto take = [&](size_t bytes) -> char* {
    char* p = ws + off;
    off += (bytes + 255) & ~(size_t)255;
    return p;
  };
  float*  dinv   = (float*)take((size_t)N * 4);                // deg then dinv (in place)
  float*  ew     = (float*)take((size_t)E * 4);                // per-edge norm
  __bf16* comb   = (__bf16*)take((size_t)N * K2 * 2);          // [N,256] bf16: [x | h]
  __bf16* cbf    = (__bf16*)take((size_t)N * H * 2);           // c input bf16
  __bf16* WhT    = (__bf16*)take((size_t)H * O * 2);           // [O,H] transposed bf16
  __bf16* WcT    = (__bf16*)take((size_t)H * O * 2);
  __bf16* WcellT = (__bf16*)take((size_t)S * K2 * C4 * 2);     // per-s [C4,K2] transposed
  float*  xw     = (float*)take((size_t)N * C4 * 4);           // GEMM result (max 512 cols)
  float*  agg    = (float*)take((size_t)N * C4 * 4);           // aggregated
  float*  hbuf   = (float*)take((size_t)N * O * 4);
  float*  cbuf   = (float*)take((size_t)N * O * 4);

  const dim3 blk(256, 1, 1);
  const int Mtiles = N / 16;  // N = 50000 -> 3125 exact

  // ---- degree / dinv / edge weights ----
  k_fill<<<g1(N), blk, 0, stream>>>(dinv, N, 1.0f);
  k_deg_accum<<<g1(E), blk, 0, stream>>>(dstIdx, dinv, E);
  k_rsqrt<<<g1(N), blk, 0, stream>>>(dinv, N);
  k_edge_w<<<g1(E), blk, 0, stream>>>(srcIdx, dstIdx, dinv, ew, E);

  // ---- bf16 conversions (weights transposed once, features once) ----
  k_pack<<<g1(NH), blk, 0, stream>>>(x, comb, NH, 0);          // comb[:, 0:128] = x
  k_cvt_bf16<<<g1(NH), blk, 0, stream>>>(c_in, cbf, NH);
  k_cvt_bf16_T<<<g1((long long)H * O), blk, 0, stream>>>(Wh, WhT, H, O, (long long)H * O);
  k_cvt_bf16_T<<<g1((long long)H * O), blk, 0, stream>>>(Wc, WcT, H, O, (long long)H * O);
  k_cvt_bf16_T<<<g1((long long)S * K2 * C4), blk, 0, stream>>>(
      Wcell, WcellT, K2, C4, (long long)S * K2 * C4);

  // ---- h_cur = GCN(x, Wh, bh) ----  (A = comb first 128 cols, lda=256)
  {
    dim3 g((unsigned)((Mtiles + 31) / 32), (unsigned)(O / 16));
    wmma_gemm_bf16<4, 4><<<g, blk, 0, stream>>>(comb, K2, WhT, H, xw, O, Mtiles);
    k_self_bias<<<g1(NH), blk, 0, stream>>>(xw, dinv, bh, hbuf, NH, O);
    long long tot = (long long)E * (O / 4);
    k_edge_scatter<<<g1(tot), blk, 0, stream>>>(srcIdx, dstIdx, ew, xw, hbuf, tot, O);
  }
  // ---- c_cur = GCN(c, Wc, bc) ----
  {
    dim3 g((unsigned)((Mtiles + 31) / 32), (unsigned)(O / 16));
    wmma_gemm_bf16<4, 4><<<g, blk, 0, stream>>>(cbf, H, WcT, H, xw, O, Mtiles);
    k_self_bias<<<g1(NH), blk, 0, stream>>>(xw, dinv, bc, cbuf, NH, O);
    long long tot = (long long)E * (O / 4);
    k_edge_scatter<<<g1(tot), blk, 0, stream>>>(srcIdx, dstIdx, ew, xw, cbuf, tot, O);
  }

  // ---- S LSTM steps ----
  for (int s = 0; s < S; ++s) {
    // combined = [x | h] -> refresh h half in bf16
    k_pack<<<g1(NH), blk, 0, stream>>>(hbuf, comb, NH, H);
    // cc = combined @ Wcell[s]  (K=256, Nc=512)
    dim3 g((unsigned)((Mtiles + 31) / 32), (unsigned)(C4 / 16));
    wmma_gemm_bf16<8, 4><<<g, blk, 0, stream>>>(
        comb, K2, WcellT + (size_t)s * K2 * C4, K2, xw, C4, Mtiles);
    // agg = dinv^2 * cc + bcell[s], then scatter over edges
    long long nC = (long long)N * C4;
    k_self_bias<<<g1(nC), blk, 0, stream>>>(xw, dinv, bcell + (size_t)s * C4, agg, nC, C4);
    long long tot = (long long)E * (C4 / 4);
    k_edge_scatter<<<g1(tot), blk, 0, stream>>>(srcIdx, dstIdx, ew, xw, agg, tot, C4);
    // gates + state update; write hs[s]
    k_lstm<<<g1(NH), blk, 0, stream>>>(agg, cbuf, hbuf, out + (size_t)s * NH, NH);
  }

  // ---- h_fin (= hs[S-1]) and c_fin ----
  k_copy<<<g1(NH), blk, 0, stream>>>(out + (size_t)(S - 1) * NH, out + (size_t)S * NH, NH);
  k_copy<<<g1(NH), blk, 0, stream>>>(cbuf, out + (size_t)(S + 1) * NH, NH);
}
